// NegativeLogLikelihoodLoss_6914897346678
// MI455X (gfx1250) — compile-verified, matrix operand order checked by reading, not run on hardware
//
#include <hip/hip_runtime.h>
#include <hip/hip_bf16.h>

typedef _Float16 v16h __attribute__((ext_vector_type(16)));
typedef float    v8f  __attribute__((ext_vector_type(8)));

#define NN 8192
#define EPSV 1e-5f

// ---------------------------------------------------------------------------
// Kernel 1: degrees via WMMA.  G is symmetric, so deg = column sums.
// D = Ones(16x32,f16) x B(32x16,f16), f32 accumulate -> every row of D holds
// the 16 partial column sums.  One block = one 16-column stripe; 4 waves
// split K=8192 into 2048-row chunks; LDS combine at the end.
// B layout used here (contiguous k; any order works as A is all ones): 16 halves per lane;
// lanes 0-15 -> K=0..15 (N=lane), lanes 16-31 -> K=16..31 (N=lane-16).
// ---------------------------------------------------------------------------
__global__ void __launch_bounds__(128)
deg_wmma_kernel(const float* __restrict__ graph, float* __restrict__ deg) {
    const int col0 = blockIdx.x * 16;
    const int wave = threadIdx.x >> 5;
    const int lane = threadIdx.x & 31;
    const int n    = lane & 15;       // column within tile
    const int kh   = lane >> 4;       // 0: K=0..15, 1: K=16..31

    v16h a;
#pragma unroll
    for (int t = 0; t < 16; ++t) a[t] = (_Float16)1.0f;

    v8f c = {0.0f, 0.0f, 0.0f, 0.0f, 0.0f, 0.0f, 0.0f, 0.0f};

    const float* colbase = graph + col0 + n;
    const int kbeg = wave * 2048;
    const int kend = kbeg + 2048;

    for (int k = kbeg; k < kend; k += 32) {
        const float* p = colbase + (size_t)(k + kh * 16) * NN;
        v16h b;
#pragma unroll
        for (int kk = 0; kk < 16; ++kk) {
            b[kk] = (_Float16)p[(size_t)kk * NN];   // exact: values are 0/1
        }
        // D = A*B + C   (emits v_wmma_f32_16x16x32_f16)
        c = __builtin_amdgcn_wmma_f32_16x16x32_f16(
                false, a, false, b, (short)0, c, false, false);
    }

    // D[0, n] lives in VGPR0 of lanes 0..15.
    __shared__ float red[4][16];
    if (lane < 16) red[wave][n] = c[0];
    __syncthreads();
    if (threadIdx.x < 16) {
        deg[col0 + threadIdx.x] = red[0][threadIdx.x] + red[1][threadIdx.x] +
                                  red[2][threadIdx.x] + red[3][threadIdx.x];
    }
}

// ---------------------------------------------------------------------------
// Kernel 2: upper-triangle log-likelihood, per-block partial sums.
// Rows i and N-1-i are paired so every (row pair) has exactly N+1 active
// elements -> perfect static load balance across the 2048 blocks.
// ---------------------------------------------------------------------------
__global__ void __launch_bounds__(256)
ll_kernel(const float* __restrict__ graph, const float* __restrict__ params,
          const float* __restrict__ deg, float* __restrict__ partials) {
    const float alpha = params[0];
    const float beta  = params[1];
    const float sigma = params[2];

    float acc = 0.0f;

    for (int ii = blockIdx.x; ii < NN / 2; ii += gridDim.x) {
#pragma unroll
        for (int half = 0; half < 2; ++half) {
            const int i = half ? (NN - 1 - ii) : ii;
            const float base = fmaf(alpha, deg[i], sigma);
            const int jstart = i & ~3;                   // keep float4 aligned
            const float* rowp = graph + (size_t)i * NN;

            for (int j = jstart + (int)threadIdx.x * 4; j < NN; j += 256 * 4) {
                const float4 g  = *reinterpret_cast<const float4*>(rowp + j);
                const float4 dd = *reinterpret_cast<const float4*>(deg + j);
                const float gv[4] = {g.x, g.y, g.z, g.w};
                const float dv[4] = {dd.x, dd.y, dd.z, dd.w};
#pragma unroll
                for (int cidx = 0; cidx < 4; ++cidx) {
                    if (j + cidx >= i) {
                        const float s = fmaf(beta, dv[cidx], base);
                        const float p = 1.0f / (1.0f + __expf(-s));
                        const float arg = (gv[cidx] == 1.0f) ? (p + EPSV)
                                                             : (1.0f - p + EPSV);
                        acc += __logf(arg);
                    }
                }
            }
        }
    }

    __shared__ float lds[256];
    lds[threadIdx.x] = acc;
    __syncthreads();
    for (int off = 128; off > 0; off >>= 1) {
        if ((int)threadIdx.x < off) lds[threadIdx.x] += lds[threadIdx.x + off];
        __syncthreads();
    }
    if (threadIdx.x == 0) partials[blockIdx.x] = lds[0];
}

// ---------------------------------------------------------------------------
// Kernel 3: deterministic final reduction, negate.
// ---------------------------------------------------------------------------
__global__ void __launch_bounds__(256)
final_reduce_kernel(const float* __restrict__ partials, float* __restrict__ out) {
    float acc = 0.0f;
    for (int idx = threadIdx.x; idx < 2048; idx += 256) acc += partials[idx];
    __shared__ float lds[256];
    lds[threadIdx.x] = acc;
    __syncthreads();
    for (int off = 128; off > 0; off >>= 1) {
        if ((int)threadIdx.x < off) lds[threadIdx.x] += lds[threadIdx.x + off];
        __syncthreads();
    }
    if (threadIdx.x == 0) out[0] = -lds[0];
}

// ---------------------------------------------------------------------------
extern "C" void kernel_launch(void* const* d_in, const int* in_sizes, int n_in,
                              void* d_out, int out_size, void* d_ws, size_t ws_size,
                              hipStream_t stream) {
    (void)in_sizes; (void)n_in; (void)out_size; (void)ws_size;

    const float* params = (const float*)d_in[0];  // (alpha, beta, sigma)
    const float* graph  = (const float*)d_in[1];  // 8192 x 8192 f32

    float* ws       = (float*)d_ws;
    float* deg      = ws;            // 8192 floats
    float* partials = ws + NN;       // 2048 floats
    float* out      = (float*)d_out; // scalar

    deg_wmma_kernel<<<NN / 16, 128, 0, stream>>>(graph, deg);
    ll_kernel<<<2048, 256, 0, stream>>>(graph, params, deg, partials);
    final_reduce_kernel<<<1, 256, 0, stream>>>(partials, out);
}
